// GraphUNetMedium_43018392436876
// MI455X (gfx1250) — matrix operand order repaired and checked by reading, not compiled, not run on hardware
//
#include <hip/hip_runtime.h>
#include <hip/hip_bf16.h>

typedef __attribute__((ext_vector_type(16))) _Float16 v16h;
typedef __attribute__((ext_vector_type(8)))  float    v8f;
typedef unsigned int u32;
typedef __attribute__((ext_vector_type(4))) u32 u32x4;
typedef __attribute__((ext_vector_type(8))) int i32x8;
typedef __attribute__((ext_vector_type(4))) int i32x4;

#define EPSN 1e-5f

#if defined(__has_builtin)
#if __has_builtin(__builtin_amdgcn_tensor_load_to_lds) && __has_builtin(__builtin_amdgcn_s_wait_tensorcnt)
#define HAVE_TDM 1
#endif
#endif
#ifndef HAVE_TDM
#define HAVE_TDM 0
#endif

// ---------------------------------------------------------------- utilities
__global__ void k_zero(float* __restrict__ p, long n) {
  long i = (long)blockIdx.x * blockDim.x + threadIdx.x;
  long stride = (long)gridDim.x * blockDim.x;
  for (; i < n; i += stride) p[i] = 0.f;
}

// ------------------------------------------------------------ instance norm
template<int C>
__global__ void __launch_bounds__(256)
k_inorm_stats(const float* __restrict__ x, int n_nodes, float* __restrict__ stats) {
  float s[C], q[C];
#pragma unroll
  for (int c = 0; c < C; c++) { s[c] = 0.f; q[c] = 0.f; }
  int stride = gridDim.x * blockDim.x;
  for (int node = blockIdx.x * blockDim.x + threadIdx.x; node < n_nodes; node += stride) {
    const float* xn = x + node * C;
#pragma unroll
    for (int c = 0; c < C; c++) { float v = xn[c]; s[c] += v; q[c] += v * v; }
  }
  __shared__ float ssum[C], ssq[C];
  for (int i = threadIdx.x; i < C; i += blockDim.x) { ssum[i] = 0.f; ssq[i] = 0.f; }
  __syncthreads();
#pragma unroll
  for (int c = 0; c < C; c++) { atomicAdd(&ssum[c], s[c]); atomicAdd(&ssq[c], q[c]); }
  __syncthreads();
  for (int c = threadIdx.x; c < C; c += blockDim.x) {
    atomicAdd(&stats[c],     ssum[c]);
    atomicAdd(&stats[C + c], ssq[c]);
  }
}

template<int C>
__global__ void k_inorm_apply(const float* __restrict__ x, float* __restrict__ y,
                              int n_nodes, const float* __restrict__ stats) {
  float mu[C], inv[C];
  float invN = 1.f / (float)n_nodes;
#pragma unroll
  for (int c = 0; c < C; c++) {
    float m = stats[c] * invN;
    float var = stats[C + c] * invN - m * m;
    mu[c] = m; inv[c] = rsqrtf(var + EPSN);
  }
  int stride = gridDim.x * blockDim.x;
  for (int node = blockIdx.x * blockDim.x + threadIdx.x; node < n_nodes; node += stride) {
    const float* xn = x + node * C;
    float* yn = y + node * C;
#pragma unroll
    for (int c = 0; c < C; c++) yn[c] = (xn[c] - mu[c]) * inv[c];
  }
}

// ------------------------------------------------- edge-weighted pool (scatter)
template<int C>
__global__ void k_pool(const float* __restrict__ x, const int* __restrict__ ei,
                       const float* __restrict__ ea, int E, float* __restrict__ out) {
  int e = blockIdx.x * blockDim.x + threadIdx.x;
  if (e >= E) return;
  int s = ei[e];
  int d = ei[E + e];
  float w = ea[e];
  const float4* xs = (const float4*)(x + s * C);
  float* od = out + d * C;
#pragma unroll
  for (int c4 = 0; c4 < C / 4; c4++) {
    float4 v = xs[c4];
    atomicAdd(od + 4 * c4 + 0, w * v.x);
    atomicAdd(od + 4 * c4 + 1, w * v.y);
    atomicAdd(od + 4 * c4 + 2, w * v.z);
    atomicAdd(od + 4 * c4 + 3, w * v.w);
  }
}

// --------------------------------------------- WMMA edge-conv (gather/GEMM/scatter)
// Full 16-edge tiles ONLY (tail edges go to k_econv_tail). One wave per tile:
// A = gathered [16 x 32k] f16 features (featA || featB || edge_attr, padded),
// B = per-fragment-swizzled f16 weight panel in LDS (staged by the Tensor Data
// Mover when available), f32 accumulate, bias+ReLU, global_atomic_add_f32 scatter.
template<int CA, int CB, int COUT, int KT, int NT>
__global__ void __launch_bounds__(256)
k_econv(const float* __restrict__ fA, const float* __restrict__ fB,
        const int* __restrict__ ei, const float* __restrict__ ea, int E,
        const float* __restrict__ W, const float* __restrict__ bias,
        float* __restrict__ out) {
  constexpr int KACT  = CA + CB + 2;
  constexpr int NFRAG = KT * NT;   // one 32x16 B fragment per (ks, nt)

  // Swizzled B panel: fragment-major, lane-major, 16 contiguous halfs per lane
  // -> each lane's fragment load is one 32-byte LDS access.
  __shared__ __align__(32) _Float16 sWsw[NFRAG * 32 * 16];
#if HAVE_TDM
  // Stage W (KACT x COUT f32) into LDS with the Tensor Data Mover.
  __shared__ float sW32[KACT * COUT];
  if (threadIdx.x < 32) {  // wave 0 issues the DMA (TDM ignores EXEC)
    unsigned long long ga = (unsigned long long)(uintptr_t)W;
    u32 lds_off = (u32)(unsigned long long)(uintptr_t)sW32;  // generic LDS addr[31:0] = offset
    u32x4 g0;
    g0[0] = 1u;                                    // count=1 (valid descriptor)
    g0[1] = lds_off;                               // lds_addr
    g0[2] = (u32)(ga & 0xffffffffull);             // global_addr[31:0]
    g0[3] = (u32)((ga >> 32) & 0x1ffffffull)       // global_addr[56:32]
          | (2u << 30);                            // type=2 ("image")
    i32x8 g1;
    g1[0] = (int)(2u << 16);                       // data_size=2 (4 bytes), mask=0
    g1[1] = (int)((u32)COUT << 16);                // tensor_dim0[15:0] @ bits 63:48
    g1[2] = (int)((u32)KACT << 16);                // dim0 hi=0 | tensor_dim1[15:0]
    g1[3] = (int)((u32)COUT << 16);                // dim1 hi=0 | tile_dim0 @ 127:112
    g1[4] = (int)(u32)KACT;                        // tile_dim1 @ 143:128, tile_dim2=0
    g1[5] = (int)(u32)COUT;                        // tensor_dim0_stride low 32
    g1[6] = 0;
    g1[7] = 0;
    i32x4 gz; gz[0] = 0; gz[1] = 0; gz[2] = 0; gz[3] = 0;
#if __clang_major__ >= 23
    i32x8 gz8;
#pragma unroll
    for (int i = 0; i < 8; i++) gz8[i] = 0;
    __builtin_amdgcn_tensor_load_to_lds(g0, g1, gz, gz, gz8, 0);
#else
    __builtin_amdgcn_tensor_load_to_lds(g0, g1, gz, gz, 0);
#endif
    __builtin_amdgcn_s_wait_tensorcnt(0);
  }
  __syncthreads();
  const float* Wsrc = sW32;
#else
  const float* Wsrc = W;
#endif
  // Build swizzled f16 panel: element i -> frag fr = i>>9, lane ln = (i>>4)&31,
  // half h = i&15 holds W[k = ks*32 + (ln>>4)*16 + h][n = nt*16 + (ln&15)].
  for (int i = threadIdx.x; i < NFRAG * 32 * 16; i += blockDim.x) {
    int h  = i & 15;
    int ln = (i >> 4) & 31;
    int fr = i >> 9;
    int ks = fr / NT, nt = fr % NT;
    int k = ks * 32 + (ln >> 4) * 16 + h;
    int n = nt * 16 + (ln & 15);
    float w = (k < KACT && n < COUT) ? Wsrc[k * COUT + n] : 0.f;
    sWsw[i] = (_Float16)w;
  }
  __syncthreads();

  int lane = threadIdx.x & 31;
  int wv   = threadIdx.x >> 5;
  int tile  = blockIdx.x * 8 + wv;
  int ebase = tile * 16;
  if (ebase + 16 > E) return;   // wave-uniform: grid covers full tiles only

  const int* dstRow = ei + E;
  int m  = lane & 15;     // A-matrix row (edge within tile) for this lane
  int hi = lane >> 4;     // half-wave selector for K layout
  int eM = ebase + m;
  int srcM = ei[eM];
  const float* rowA = fA + srcM * CA;
  const float* rowB = (CB > 0) ? (fB + srcM * CB) : fA;
  const float* rowE = ea + eM * 2;

  // 8 destination indices this lane's accumulator rows cover (contiguous, aligned)
  const int* dp = dstRow + ebase + 8 * hi;
  int4 dv0 = *(const int4*)dp;
  int4 dv1 = *(const int4*)(dp + 4);
  int dsts[8] = {dv0.x, dv0.y, dv0.z, dv0.w, dv1.x, dv1.y, dv1.z, dv1.w};

  v8f acc[NT];
#pragma unroll
  for (int t = 0; t < NT; t++)
#pragma unroll
    for (int r = 0; r < 8; r++) acc[t][r] = 0.f;

  const v16h* Bp = (const v16h*)sWsw;

#pragma unroll
  for (int ks = 0; ks < KT; ks++) {
    // ---- A fragment: 16x32 f16, contiguous K order (matches B panel)
    //  lanes 0-15:  VGPR0-7 hold K=0..15
    //  lanes 16-31: VGPR0-7 hold K=16..31
    v16h a;
#pragma unroll
    for (int h = 0; h < 16; h++) {
      int k  = h + (hi ? 16 : 0);
      int kg = ks * 32 + k;
      float f = 0.f;
      if (kg < KACT) {
        if (kg < CA)           f = rowA[kg];
        else if (kg < CA + CB) f = rowB[kg - CA];
        else                   f = rowE[kg - CA - CB];
      }
      a[h] = (_Float16)f;
    }
#pragma unroll
    for (int nt = 0; nt < NT; nt++) {
      v16h b = Bp[(ks * NT + nt) * 32 + lane];
      acc[nt] = __builtin_amdgcn_wmma_f32_16x16x32_f16(
          /*neg_a=*/false, a, /*neg_b=*/false, b,
          /*c_mod=*/(short)0, acc[nt], /*reuse_a=*/false, /*reuse_b=*/false);
    }
  }

  // ---- bias + ReLU + scatter-add. C/D layout: VGPR r -> M=r+8*hi, N=lane&15.
#pragma unroll
  for (int nt = 0; nt < NT; nt++) {
    int n = nt * 16 + m;
    if ((COUT % 16 == 0) || (n < COUT)) {   // folds away when COUT is 16-aligned
      float bb = bias[n];
#pragma unroll
      for (int r = 0; r < 8; r++) {
        float v = acc[nt][r] + bb;
        v = v > 0.f ? v : 0.f;
        atomicAdd(out + (dsts[r] * COUT + n), v);
      }
    }
  }
}

// ---- scalar tail for the (rare) E % 16 leftover edges
template<int CA, int CB, int COUT>
__global__ void k_econv_tail(const float* __restrict__ fA, const float* __restrict__ fB,
                             const int* __restrict__ ei, const float* __restrict__ ea,
                             int E, int e0, const float* __restrict__ W,
                             const float* __restrict__ bias, float* __restrict__ out) {
  int e = e0 + blockIdx.x * blockDim.x + threadIdx.x;
  if (e >= E) return;
  int s = ei[e], d = ei[E + e];
  constexpr int K = CA + CB + 2;
#pragma clang loop unroll(disable)
  for (int n = 0; n < COUT; n++) {
    float accv = bias[n];
#pragma clang loop unroll(disable)
    for (int k = 0; k < K; k++) {
      float f = (k < CA) ? fA[s * CA + k]
              : (k < CA + CB) ? fB[s * CB + (k - CA)]
              : ea[2 * e + (k - CA - CB)];
      accv += f * W[k * COUT + n];
    }
    accv = accv > 0.f ? accv : 0.f;
    atomicAdd(out + (d * COUT + n), accv);
  }
}

// --------------------------------------------------------- final 24 -> 1 linear
__global__ void k_final(const float* __restrict__ x, const float* __restrict__ w,
                        const float* __restrict__ b, float* __restrict__ out, int n) {
  int i = blockIdx.x * blockDim.x + threadIdx.x;
  if (i >= n) return;
  const float* xi = x + i * 24;
  float s = b[0];
#pragma unroll
  for (int c = 0; c < 24; c++) s += xi[c] * w[c];
  out[i] = s;
}

// ================================================================ host side
static inline int divup_i(long a, long b) { return (int)((a + b - 1) / b); }
static inline long lmin(long a, long b) { return a < b ? a : b; }

static void zero_buf(float* p, long n, hipStream_t s) {
  int blocks = (int)lmin(4096, (n + 255) / 256);
  k_zero<<<blocks, 256, 0, s>>>(p, n);
}

template<int C>
static void inorm(const float* x, float* y, long n, float* stats, hipStream_t s) {
  k_zero<<<1, 64, 0, s>>>(stats, 2 * C);
  int blocks = (int)lmin(1024, (n + 255) / 256);
  k_inorm_stats<C><<<blocks, 256, 0, s>>>(x, (int)n, stats);
  k_inorm_apply<C><<<divup_i(n, 256), 256, 0, s>>>(x, y, (int)n, stats);
}

template<int C>
static void pool(const float* x, const int* ei, const float* ea, long E,
                 float* out, long n_out, hipStream_t s) {
  zero_buf(out, n_out * (long)C, s);
  k_pool<C><<<divup_i(E, 256), 256, 0, s>>>(x, ei, ea, (int)E, out);
}

template<int CA, int CB, int COUT, int KT, int NT>
static void econv(const float* fA, const float* fB, const int* ei, const float* ea,
                  long E, const float* W, const float* b, float* out, long n_out,
                  hipStream_t s) {
  zero_buf(out, n_out * (long)COUT, s);
  long fullTiles = E / 16;
  if (fullTiles > 0) {
    int blocks = (int)((fullTiles + 7) / 8);  // 8 waves (tiles) per 256-thread block
    k_econv<CA, CB, COUT, KT, NT><<<blocks, 256, 0, s>>>(fA, fB, ei, ea, (int)E, W, b, out);
  }
  if (E % 16) {
    k_econv_tail<CA, CB, COUT><<<1, 16, 0, s>>>(fA, fB, ei, ea, (int)E,
                                                (int)(fullTiles * 16), W, b, out);
  }
}

extern "C" void kernel_launch(void* const* d_in, const int* in_sizes, int n_in,
                              void* d_out, int out_size, void* d_ws, size_t ws_size,
                              hipStream_t stream) {
  constexpr long NC = 400000, NF = 800000;
  constexpr long NP0 = 400000, NP1 = 100000, NP2 = 25000, NP3 = 6250, NP4 = 1600;

  const float* xcc  = (const float*)d_in[0];
  const float* xf   = (const float*)d_in[1];
  const int*   cf_ei = (const int*)d_in[2];  const float* cf_ea = (const float*)d_in[3];
  const int*   fp_ei = (const int*)d_in[4];  const float* fp_ea = (const float*)d_in[5];
  const int*   pc_ei = (const int*)d_in[6];  const float* pc_ea = (const float*)d_in[7];
  const int*   pp_ei[5]; const float* pp_ea[5];
  for (int l = 0; l < 5; l++) { pp_ei[l] = (const int*)d_in[8 + 2*l]; pp_ea[l] = (const float*)d_in[9 + 2*l]; }
  // interleaved: pool_l ei/ea, unpool_l ei/ea
  const int* pool_ei[4]; const float* pool_ea[4];
  const int* unpool_ei[4]; const float* unpool_ea[4];
  for (int l = 0; l < 4; l++) {
    pool_ei[l]   = (const int*)d_in[18 + 4*l];   pool_ea[l]   = (const float*)d_in[19 + 4*l];
    unpool_ei[l] = (const int*)d_in[20 + 4*l];   unpool_ea[l] = (const float*)d_in[21 + 4*l];
  }
  const float* W_cf = (const float*)d_in[34]; const float* b_cf = (const float*)d_in[35];
  const float* W_fp = (const float*)d_in[36]; const float* b_fp = (const float*)d_in[37];
  const float* W2   = (const float*)d_in[38]; const float* b2   = (const float*)d_in[39];
  const float* W3   = (const float*)d_in[40]; const float* b3   = (const float*)d_in[41];
  const float* W4   = (const float*)d_in[42]; const float* b4   = (const float*)d_in[43];
  const float* W5a  = (const float*)d_in[44]; const float* b5a  = (const float*)d_in[45];
  const float* W5b  = (const float*)d_in[46]; const float* b5b  = (const float*)d_in[47];
  const float* W6   = (const float*)d_in[48]; const float* b6   = (const float*)d_in[49];
  const float* W7   = (const float*)d_in[50]; const float* b7   = (const float*)d_in[51];
  const float* W8   = (const float*)d_in[52]; const float* b8   = (const float*)d_in[53];
  const float* W9   = (const float*)d_in[54]; const float* b9   = (const float*)d_in[55];
  const float* W_pc = (const float*)d_in[56]; const float* b_pc = (const float*)d_in[57];
  const float* W_fin= (const float*)d_in[58]; const float* b_fin= (const float*)d_in[59];

  long E_cf = in_sizes[2] / 2, E_fp = in_sizes[4] / 2, E_pc = in_sizes[6] / 2;
  long E_pp[5], E_pool[4], E_unpool[4];
  for (int l = 0; l < 5; l++) E_pp[l] = in_sizes[8 + 2*l] / 2;
  for (int l = 0; l < 4; l++) { E_pool[l] = in_sizes[18 + 4*l] / 2; E_unpool[l] = in_sizes[20 + 4*l] / 2; }

  // ---- workspace layout (f32 elements)
  float* ws = (float*)d_ws;
  long off = 0;
  auto alloc = [&](long n) { float* p = ws + off; off += n; return p; };
  float* xccn = alloc(NC * 2);
  float* xfn  = alloc(NF * 4);
  float* t    = alloc(NF * 20);
  float* c1   = alloc(NP0 * 24);
  float* p1   = alloc(NP1 * 24);
  float* c2   = alloc(NP1 * 24);
  float* p2   = alloc(NP2 * 24);
  float* c3   = alloc(NP2 * 24);
  float* p3   = alloc(NP3 * 24);
  float* c4   = alloc(NP3 * 48);
  float* p4   = alloc(NP4 * 48);
  float* z4a  = alloc(NP4 * 48);
  float* z4b  = alloc(NP4 * 48);
  float* c5   = alloc(NP3 * 48);
  float* t3   = alloc(NP3 * 48);
  float* c6   = alloc(NP2 * 48);
  float* t2   = alloc(NP2 * 48);
  float* c7   = alloc(NP1 * 48);
  float* t1   = alloc(NP1 * 24);
  float* c8   = alloc(NP0 * 24);
  float* z9   = alloc(NP0 * 24);
  float* c9   = alloc(NC * 24);
  float* stats = alloc(128);
  (void)ws_size; (void)n_in; (void)out_size;

  // ---- pipeline
  inorm<2>(xcc, xccn, NC, stats, stream);
  inorm<4>(xf,  xfn,  NF, stats, stream);

  // ConvBlock1: cells->faces (4->20), then [t,xf]->points (26->24)
  econv<2, 0, 20, 1, 2>(xccn, nullptr, cf_ei, cf_ea, E_cf, W_cf, b_cf, t,  NF,  stream);
  econv<20, 4, 24, 1, 2>(t, xfn,       fp_ei, fp_ea, E_fp, W_fp, b_fp, c1, NP0, stream);
  inorm<24>(c1, c1, NP0, stats, stream);

  // encoder
  pool<24>(c1, pool_ei[0], pool_ea[0], E_pool[0], p1, NP1, stream);
  econv<24, 0, 24, 1, 2>(p1, nullptr, pp_ei[1], pp_ea[1], E_pp[1], W2, b2, c2, NP1, stream);
  inorm<24>(c2, c2, NP1, stats, stream);

  pool<24>(c2, pool_ei[1], pool_ea[1], E_pool[1], p2, NP2, stream);
  econv<24, 0, 24, 1, 2>(p2, nullptr, pp_ei[2], pp_ea[2], E_pp[2], W3, b3, c3, NP2, stream);
  inorm<24>(c3, c3, NP2, stats, stream);

  pool<24>(c3, pool_ei[2], pool_ea[2], E_pool[2], p3, NP3, stream);
  econv<24, 0, 48, 1, 3>(p3, nullptr, pp_ei[3], pp_ea[3], E_pp[3], W4, b4, c4, NP3, stream);
  inorm<48>(c4, c4, NP3, stats, stream);

  // bottleneck
  pool<48>(c4, pool_ei[3], pool_ea[3], E_pool[3], p4, NP4, stream);
  econv<48, 0, 48, 2, 3>(p4,  nullptr, pp_ei[4], pp_ea[4], E_pp[4], W5a, b5a, z4a, NP4, stream);
  econv<48, 0, 48, 2, 3>(z4a, nullptr, pp_ei[4], pp_ea[4], E_pp[4], W5b, b5b, z4b, NP4, stream);
  pool<48>(z4b, unpool_ei[3], unpool_ea[3], E_unpool[3], c5, NP3, stream);
  inorm<48>(c5, c5, NP3, stats, stream);

  // decoder
  econv<48, 48, 48, 4, 3>(c5, c4, pp_ei[3], pp_ea[3], E_pp[3], W6, b6, t3, NP3, stream);
  pool<48>(t3, unpool_ei[2], unpool_ea[2], E_unpool[2], c6, NP2, stream);
  inorm<48>(c6, c6, NP2, stats, stream);

  econv<48, 24, 48, 3, 3>(c6, c3, pp_ei[2], pp_ea[2], E_pp[2], W7, b7, t2, NP2, stream);
  pool<48>(t2, unpool_ei[1], unpool_ea[1], E_unpool[1], c7, NP1, stream);
  inorm<48>(c7, c7, NP1, stats, stream);

  econv<48, 24, 24, 3, 2>(c7, c2, pp_ei[1], pp_ea[1], E_pp[1], W8, b8, t1, NP1, stream);
  pool<24>(t1, unpool_ei[0], unpool_ea[0], E_unpool[0], c8, NP0, stream);
  inorm<24>(c8, c8, NP0, stats, stream);

  // ConvBlock9: pointPoint (50->24) then points->cells (26->24)
  econv<24, 24, 24, 2, 2>(c8, c1, pp_ei[0], pp_ea[0], E_pp[0], W9, b9, z9, NP0, stream);
  econv<24, 0, 24, 1, 2>(z9, nullptr, pc_ei, pc_ea, E_pc, W_pc, b_pc, c9, NC, stream);
  inorm<24>(c9, c9, NC, stats, stream);

  // final per-node linear 24 -> 1
  k_final<<<divup_i(NC, 256), 256, 0, stream>>>(c9, W_fin, b_fin, (float*)d_out, NC);
}
